// GatedBimodal_72335839199274
// MI455X (gfx1250) — matrix operand order repaired and checked by reading, not compiled, not run on hardware
//
#include <hip/hip_runtime.h>
#include <hip/hip_bf16.h>
#include <stdint.h>

// ---------------------------------------------------------------------------
// GatedBimodal fused kernel for MI455X (gfx1250, wave32, WMMA)
//   hv = tanh(x1 @ Wv^T); ht = tanh(x2 @ Wt^T)
//   z  = sigmoid([x1|x2] @ Wz)  -- bf16 WMMA GEMM, M=16384, K=2048, N=1024
//   fused = z*hv + (1-z)*ht
// d_out = [fused (B*D) | z (B*D)] fp32
// d_ws  = [hv (64KB) | ht (64KB) | WzT bf16 [N][K] (4MB)]
// ---------------------------------------------------------------------------

#define B_ROWS 16384
#define DDIM   1024
#define KDIM   2048

typedef __attribute__((ext_vector_type(16))) __bf16 v16bf;
typedef __attribute__((ext_vector_type(8)))  __bf16 v8bf;
typedef __attribute__((ext_vector_type(8)))  float  v8f;
typedef __attribute__((ext_vector_type(4)))  int    v4i;

// ---- gfx1250 async global->LDS copy (ASYNCcnt path), with safe fallback ----
#if defined(__has_builtin)
#if __has_builtin(__builtin_amdgcn_global_load_async_to_lds_b128) && \
    __has_builtin(__builtin_amdgcn_s_wait_asynccnt)
#define HAVE_ASYNC_LDS 1
#endif
#endif
#ifndef HAVE_ASYNC_LDS
#define HAVE_ASYNC_LDS 0
#endif

typedef __attribute__((address_space(1))) v4i as1_v4i;  // global 16B chunk
typedef __attribute__((address_space(3))) v4i as3_v4i;  // LDS 16B chunk

__device__ __forceinline__ void copy16_g2l(const void* g, void* l) {
#if HAVE_ASYNC_LDS
  __builtin_amdgcn_global_load_async_to_lds_b128((as1_v4i*)g, (as3_v4i*)l,
                                                 0, 0);
#else
  uint4 v = *(const uint4*)g;
  *(uint4*)l = v;
#endif
}

__device__ __forceinline__ void wait_async_stage() {
#if HAVE_ASYNC_LDS
  __builtin_amdgcn_s_wait_asynccnt(0);
#endif
}

// --------------------------- hv / ht kernel --------------------------------
__global__ void rowdot_tanh_kernel(const float* __restrict__ x,
                                   const float* __restrict__ w,
                                   float* __restrict__ out) {
  const int wave = threadIdx.x >> 5;
  const int lane = threadIdx.x & 31;
  const int row  = blockIdx.x * 8 + wave;
  const float* xr = x + (size_t)row * DDIM;

  float sum = 0.0f;
#pragma unroll 4
  for (int i = lane; i < DDIM; i += 32) sum += xr[i] * w[i];
#pragma unroll
  for (int off = 16; off > 0; off >>= 1) sum += __shfl_xor(sum, off, 32);
  if (lane == 0) out[row] = tanhf(sum);
}

// --------------- Wz fp32 [K][N] -> bf16 transposed WzT [N][K] --------------
// Thread: one n, 8 consecutive k. Reads coalesced along n; 16B writes.
__global__ void convert_transpose_wz(const float* __restrict__ Wz,
                                     __bf16* __restrict__ WzT) {
  const int g  = blockIdx.x * blockDim.x + threadIdx.x;
  const int n  = g & (DDIM - 1);
  const int k0 = (g >> 10) * 8;
  v8bf o;
#pragma unroll
  for (int j = 0; j < 8; ++j)
    o[j] = (__bf16)Wz[(size_t)(k0 + j) * DDIM + n];
  *(v8bf*)(WzT + (size_t)n * KDIM + k0) = o;
}

// ----------------------------- GEMM kernel ---------------------------------
// Block: 256 threads (8 waves). Block tile: 128(M) x 128(N). K-step: 64.
__global__ void __launch_bounds__(256)
gated_gemm_wmma_kernel(const float* __restrict__ x1,
                       const float* __restrict__ x2,
                       const __bf16* __restrict__ WzT,
                       const float* __restrict__ hv,
                       const float* __restrict__ ht,
                       float* __restrict__ fused,
                       float* __restrict__ z_out) {
  __shared__ __bf16 As[128 * 64];   // [m][k], row stride 64
  __shared__ __bf16 Bs[128 * 64];   // [n][k], row stride 64

  const int tid  = threadIdx.x;
  const int wave = tid >> 5;
  const int lane = tid & 31;
  const int half = lane >> 4;
  const int l16  = lane & 15;

  const int n0 = blockIdx.x * 128;
  const int m0 = blockIdx.y * 128;
  const int wm = wave >> 1;         // 4 waves along M
  const int wn = wave & 1;          // 2 waves along N

  // staging assignments
  const int aRow = tid >> 1;        // 0..127
  const int aKh  = (tid & 1) * 32;  // 0 or 32
  const int bRow = tid >> 1;        // 0..127 (n)
  const int bKo  = (tid & 1) * 32;  // element offset in 64-k row

  v8f acc[2][4];
#pragma unroll
  for (int i = 0; i < 2; ++i)
#pragma unroll
    for (int j = 0; j < 4; ++j) acc[i][j] = (v8f){};

  for (int kk = 0; kk < KDIM; kk += 64) {
    // ---- stage B: WzT tile (already bf16, already [n][k]) -> LDS, async
    {
      const __bf16* gB = WzT + (size_t)(n0 + bRow) * KDIM + kk + bKo;
      __bf16* lB = Bs + bRow * 64 + bKo;
#pragma unroll
      for (int s = 0; s < 4; ++s)            // 4 x 16B = 32 bf16 per thread
        copy16_g2l(gB + s * 8, lB + s * 8);
    }
    // ---- stage A: rows m0..+127, cols kk..+63 of [x1|x2], fp32 -> bf16
    {
      const float* srcBase = (kk < DDIM) ? x1 : x2;
      const int    kcol    = (kk < DDIM) ? kk : (kk - DDIM);
      const float* p = srcBase + (size_t)(m0 + aRow) * DDIM + kcol + aKh;
      if (kk + 64 < KDIM) {
        const float* nsrc = (kk + 64 < DDIM) ? x1 : x2;
        const int    nk   = (kk + 64 < DDIM) ? (kk + 64) : (kk + 64 - DDIM);
        __builtin_prefetch(nsrc + (size_t)(m0 + aRow) * DDIM + nk + aKh, 0, 1);
      }
      const float4* p4 = (const float4*)p;
      __bf16* dst = As + aRow * 64 + aKh;
#pragma unroll
      for (int g = 0; g < 4; ++g) {
        float4 u = p4[g * 2];
        float4 v = p4[g * 2 + 1];
        v8bf o = {(__bf16)u.x, (__bf16)u.y, (__bf16)u.z, (__bf16)u.w,
                  (__bf16)v.x, (__bf16)v.y, (__bf16)v.z, (__bf16)v.w};
        *(v8bf*)(dst + g * 8) = o;
      }
    }
    wait_async_stage();
    __syncthreads();

    // ---- compute: two 32-K WMMA steps over the 64-K LDS stage
#pragma unroll
    for (int ks = 0; ks < 64; ks += 32) {
      v16bf aFrag[2];
#pragma unroll
      for (int mt = 0; mt < 2; ++mt) {
        const __bf16* ap = As + (wm * 32 + mt * 16 + l16) * 64 + ks + half * 8;
        v8bf lo = *(const v8bf*)(ap);
        v8bf hi = *(const v8bf*)(ap + 16);
        aFrag[mt] = __builtin_shufflevector(lo, hi, 0, 1, 2, 3, 4, 5, 6, 7,
                                            8, 9, 10, 11, 12, 13, 14, 15);
      }
      v16bf bFrag[4];
#pragma unroll
      for (int nt = 0; nt < 4; ++nt) {
        const __bf16* bp = Bs + (wn * 64 + nt * 16 + l16) * 64 + ks + half * 8;
        v8bf lo = *(const v8bf*)(bp);
        v8bf hi = *(const v8bf*)(bp + 16);
        bFrag[nt] = __builtin_shufflevector(lo, hi, 0, 1, 2, 3, 4, 5, 6, 7,
                                            8, 9, 10, 11, 12, 13, 14, 15);
      }
#pragma unroll
      for (int mt = 0; mt < 2; ++mt)
#pragma unroll
        for (int nt = 0; nt < 4; ++nt)
          acc[mt][nt] = __builtin_amdgcn_wmma_f32_16x16x32_bf16(
              false, aFrag[mt], false, bFrag[nt], (short)0, acc[mt][nt],
              false, false);
    }
    __syncthreads();
  }

  // ---- epilogue: sigmoid + gating + store (D layout: VGPR r -> row r+8*half)
#pragma unroll
  for (int mt = 0; mt < 2; ++mt) {
    const int gmBase = m0 + wm * 32 + mt * 16 + half * 8;
#pragma unroll
    for (int nt = 0; nt < 4; ++nt) {
      const int gn = n0 + wn * 64 + nt * 16 + l16;
#pragma unroll
      for (int r = 0; r < 8; ++r) {
        const int gm = gmBase + r;
        float v  = acc[mt][nt][r];
        float zz = 1.0f / (1.0f + __expf(-v));
        float f  = zz * hv[gm] + (1.0f - zz) * ht[gm];
        const size_t idx = (size_t)gm * DDIM + gn;
        fused[idx] = f;
        z_out[idx] = zz;
      }
    }
  }
}

// ------------------------------- launcher ----------------------------------
extern "C" void kernel_launch(void* const* d_in, const int* in_sizes, int n_in,
                              void* d_out, int out_size, void* d_ws, size_t ws_size,
                              hipStream_t stream) {
  const float* x1 = (const float*)d_in[0];
  const float* x2 = (const float*)d_in[1];
  const float* Wz = (const float*)d_in[2];
  const float* Wv = (const float*)d_in[3];
  const float* Wt = (const float*)d_in[4];

  float*  hv  = (float*)d_ws;                                  // 64 KB
  float*  ht  = hv + B_ROWS;                                   // 64 KB
  __bf16* WzT = (__bf16*)((char*)d_ws + 2 * B_ROWS * sizeof(float)); // 4 MB

  float* fused = (float*)d_out;
  float* z_out = fused + (size_t)B_ROWS * DDIM;

  rowdot_tanh_kernel<<<B_ROWS / 8, 256, 0, stream>>>(x1, Wv, hv);
  rowdot_tanh_kernel<<<B_ROWS / 8, 256, 0, stream>>>(x2, Wt, ht);

  // one-time Wz fp32[K][N] -> bf16 WzT[N][K] (resident in 192MB L2 afterwards)
  convert_transpose_wz<<<(DDIM * (KDIM / 8)) / 256, 256, 0, stream>>>(Wz, WzT);

  dim3 grid(DDIM / 128, B_ROWS / 128);  // (8, 128)
  gated_gemm_wmma_kernel<<<grid, 256, 0, stream>>>(x1, x2, WzT, hv, ht,
                                                   fused, z_out);
}
